// Backbone_12661563589002
// MI455X (gfx1250) — matrix operand order repaired and checked by reading, not compiled, not run on hardware
//
#include <hip/hip_runtime.h>
#include <stdint.h>

// ============================================================================
// MI455X (gfx1250) implementation of conv backbone + MoE embedding head.
//
// Design (compile-only reasoning, see analysis):
//  - conv1/conv2 as implicit-GEMM via v_wmma_f32_16x16x32_bf16
//    (M = output pixels, N = out-channels, K = Cin*25 zero-padded to mult of 32)
//  - weights pre-packed to [Cout][Kpad] bf16 so WMMA B-fragments are
//    contiguous 16-element K runs per lane; im2col A tiles staged in LDS.
//  - BN batch stats via deterministic two-stage reduction (no float atomics).
//  - expert MLP as WMMA GEMMs over bf16 with fused bias/ReLU epilogues.
//  - workspace use ~365 MB (c1out/c2out/h1 in bf16) — all ws buffers are
//    written before read every call; no state carried across calls.
// ============================================================================

typedef __attribute__((ext_vector_type(16))) __bf16       v16bf;
typedef __attribute__((ext_vector_type(8)))  float        v8f;
typedef __attribute__((ext_vector_type(8)))  unsigned int v8u;
typedef unsigned short bf_t;

__device__ __forceinline__ bf_t f2bf(float f) {
  unsigned int u = __float_as_uint(f);
  u += 0x7FFFu + ((u >> 16) & 1u);          // round-to-nearest-even
  return (bf_t)(u >> 16);
}
__device__ __forceinline__ float bf2f(bf_t h) {
  return __uint_as_float(((unsigned int)h) << 16);
}
__device__ __forceinline__ v8f vzero8() {
  v8f z = {0.f, 0.f, 0.f, 0.f, 0.f, 0.f, 0.f, 0.f};
  return z;
}
__device__ __forceinline__ v8f wmma_bf16(v8u a, v8u b, v8f c) {
  return __builtin_amdgcn_wmma_f32_16x16x32_bf16(
      false, __builtin_bit_cast(v16bf, a),
      false, __builtin_bit_cast(v16bf, b),
      (short)0, c, false, false);
}

// ---------------------------------------------------------------------------
// Weight prep: [O,K] f32 -> [O,KP] bf16, zero padded in K.
// ---------------------------------------------------------------------------
__global__ void pack_pad_bf16(const float* __restrict__ src, bf_t* __restrict__ dst,
                              int O, int K, int KP) {
  int i = blockIdx.x * blockDim.x + threadIdx.x;
  if (i >= O * KP) return;
  int o = i / KP, k = i - o * KP;
  dst[i] = (k < K) ? f2bf(src[o * K + k]) : (bf_t)0;
}

// [E,R,C] f32 -> [E,C,R] bf16
__global__ void transpose_to_bf16(const float* __restrict__ src, bf_t* __restrict__ dst,
                                  int E, int R, int C) {
  int i = blockIdx.x * blockDim.x + threadIdx.x;
  if (i >= E * R * C) return;
  int e = i / (R * C);
  int rem = i - e * (R * C);
  int c = rem / R, r = rem - c * R;
  dst[i] = f2bf(src[((size_t)e * R + r) * C + c]);
}

// ---------------------------------------------------------------------------
// Implicit-GEMM 5x5 pad=2 conv with WMMA bf16.
//  in:  [B, CIN, IMG, IMG]  (f32 if IN_F32 else bf16)
//  wt:  [NT*16, KSTEPS*32] bf16 (K = ci*25 + fy*5 + fx, zero padded)
//  out: [B, NT*16, IMG, IMG] bf16, conv bias folded in.
// grid: (IMG*IMG/64, B), block 128 (4 waves). Wave w does rows w*16..w*16+15.
// ---------------------------------------------------------------------------
template <int CIN, int KSTEPS, int IMG, int NT, bool IN_F32>
__global__ __launch_bounds__(128) void conv5x5_wmma(const void* __restrict__ in_,
                                                    const bf_t* __restrict__ wt,
                                                    const float* __restrict__ bias,
                                                    bf_t* __restrict__ out) {
  constexpr int P    = IMG * IMG;
  constexpr int COUT = NT * 16;
  constexpr int KW   = KSTEPS * 32;

  __shared__ alignas(16) bf_t smA[64 * 32];   // im2col tile: 64 pixels x 32 k

  const int tid  = threadIdx.x;
  const int wv   = tid >> 5;
  const int lane = tid & 31;
  const int g    = lane >> 4;
  const int hs   = lane & 15;
  const int b    = blockIdx.y;
  const int pixbase = blockIdx.x * 64;

  const float* inF = (const float*)in_;
  const bf_t*  inH = (const bf_t*)in_;

  v8f acc[NT];
#pragma unroll
  for (int t = 0; t < NT; ++t) acc[t] = vzero8();

  for (int ks = 0; ks < KSTEPS; ++ks) {
    // cooperative im2col fill (bounds-checked gather, zero pad)
    for (int i = tid; i < 64 * 32; i += 128) {
      int mloc = i >> 5, kl = i & 31;
      int k  = ks * 32 + kl;
      int ci = k / 25;
      int r  = k - ci * 25;
      int fy = r / 5, fx = r - fy * 5;
      int pix = pixbase + mloc;
      int y = pix / IMG, x = pix - y * IMG;
      int iy = y + fy - 2, ix = x + fx - 2;
      bf_t v = 0;
      if (ci < CIN && (unsigned)iy < (unsigned)IMG && (unsigned)ix < (unsigned)IMG) {
        size_t idx = ((size_t)(b * CIN + ci) * IMG + iy) * IMG + ix;
        v = IN_F32 ? f2bf(inF[idx]) : inH[idx];
      }
      smA[i] = v;
    }
    __syncthreads();

    // A fragment: row m = hs of this wave's tile; two 8-elem K runs per lane
    const unsigned int* pa = (const unsigned int*)&smA[(wv * 16 + hs) * 32];
    const int o0 = g * 8, o1 = g * 8 + 4;
    v8u ap;
    ap[0] = pa[o0];     ap[1] = pa[o0 + 1]; ap[2] = pa[o0 + 2]; ap[3] = pa[o0 + 3];
    ap[4] = pa[o1];     ap[5] = pa[o1 + 1]; ap[6] = pa[o1 + 2]; ap[7] = pa[o1 + 3];

#pragma unroll
    for (int t = 0; t < NT; ++t) {
      const unsigned int* pb =
          (const unsigned int*)&wt[(size_t)(t * 16 + hs) * KW + ks * 32 + g * 16];
      v8u bp;
#pragma unroll
      for (int j = 0; j < 8; ++j) bp[j] = pb[j];
      acc[t] = wmma_bf16(ap, bp, acc[t]);
    }
    __syncthreads();
  }

  // epilogue: D lane (g,hs) holds N=hs, M=g*8+r ; add conv bias, store bf16
#pragma unroll
  for (int t = 0; t < NT; ++t) {
    int co = t * 16 + hs;
    float bc = bias[co];
#pragma unroll
    for (int r = 0; r < 8; ++r) {
      int pix = pixbase + wv * 16 + g * 8 + r;
      out[(size_t)(b * COUT + co) * P + pix] = f2bf(acc[t][r] + bc);
    }
  }
}

// ---------------------------------------------------------------------------
// Per-channel sum / sumsq partials over [B, C, P] bf16. Deterministic:
// partial per (c, blockIdx.y), final serial sum in bn_finalize.
// ---------------------------------------------------------------------------
__global__ __launch_bounds__(256) void chan_stats(const bf_t* __restrict__ data,
                                                  float* __restrict__ psum,
                                                  float* __restrict__ psq,
                                                  int C, int P, int nBP) {
  int c  = blockIdx.x;
  int nb = gridDim.y;
  long stride = (long)nb * 256;
  float s = 0.f, q = 0.f;
  for (long i = (long)blockIdx.y * 256 + threadIdx.x; i < (long)nBP; i += stride) {
    long bb = i / P;
    long p  = i - bb * P;
    float v = bf2f(data[((size_t)bb * C + c) * P + p]);
    s += v;
    q += v * v;
  }
  __shared__ float rs[256], rq[256];
  rs[threadIdx.x] = s;
  rq[threadIdx.x] = q;
  __syncthreads();
  for (int o = 128; o > 0; o >>= 1) {
    if (threadIdx.x < o) {
      rs[threadIdx.x] += rs[threadIdx.x + o];
      rq[threadIdx.x] += rq[threadIdx.x + o];
    }
    __syncthreads();
  }
  if (threadIdx.x == 0) {
    psum[c * nb + blockIdx.y] = rs[0];
    psq[c * nb + blockIdx.y]  = rq[0];
  }
}

__global__ void bn_finalize(const float* __restrict__ psum, const float* __restrict__ psq,
                            int nPart, const float* __restrict__ gam,
                            const float* __restrict__ bet, float* __restrict__ ab,
                            int C, float invN) {
  int c = threadIdx.x;
  if (c >= C) return;
  float s = 0.f, q = 0.f;
  for (int i = 0; i < nPart; ++i) { s += psum[c * nPart + i]; q += psq[c * nPart + i]; }
  float mean = s * invN;
  float var  = q * invN - mean * mean;
  float sc   = gam[c] * rsqrtf(var + 1e-5f);
  ab[c]     = sc;
  ab[C + c] = bet[c] - mean * sc;
}

// BN + ReLU + 2x2 maxpool: in [B,C,IH,IH] bf16 -> out [B,C,IH/2,IH/2] bf16
__global__ void bnpool(const bf_t* __restrict__ in, const float* __restrict__ ab,
                       bf_t* __restrict__ out, int C, int IH, int n) {
  int i = blockIdx.x * blockDim.x + threadIdx.x;
  if (i >= n) return;
  int OH = IH >> 1;
  int ow = i % OH; int t = i / OH;
  int oh = t % OH; t /= OH;
  int c  = t % C;  int b = t / C;
  float s = ab[c], sh = ab[C + c];
  size_t base = ((size_t)(b * C + c) * IH + oh * 2) * IH + ow * 2;
  float v0 = s * bf2f(in[base])          + sh;
  float v1 = s * bf2f(in[base + 1])      + sh;
  float v2 = s * bf2f(in[base + IH])     + sh;
  float v3 = s * bf2f(in[base + IH + 1]) + sh;
  float m = fmaxf(fmaxf(v0, v1), fmaxf(v2, v3));
  out[i] = f2bf(fmaxf(m, 0.f));
}

// BN + ReLU + 2x2 maxpool + 4x4 avgpool fused:
// c2out [512,32,56,56] bf16 -> feat [512,1568] f32 (+ bf16 copy)
__global__ void bnpoolavg(const bf_t* __restrict__ in, const float* __restrict__ ab,
                          float* __restrict__ feat, bf_t* __restrict__ featbf) {
  int i = blockIdx.x * blockDim.x + threadIdx.x;
  if (i >= 512 * 32 * 49) return;
  int ox = i % 7; int t = i / 7;
  int oy = t % 7; t /= 7;
  int c  = t % 32; int b = t / 32;
  float s = ab[c], sh = ab[32 + c];
  float acc = 0.f;
#pragma unroll
  for (int wy = 0; wy < 4; ++wy) {
#pragma unroll
    for (int wx = 0; wx < 4; ++wx) {
      int iy = oy * 8 + wy * 2, ix = ox * 8 + wx * 2;
      size_t base = ((size_t)(b * 32 + c) * 56 + iy) * 56 + ix;
      float v0 = s * bf2f(in[base])          + sh;
      float v1 = s * bf2f(in[base + 1])      + sh;
      float v2 = s * bf2f(in[base + 56])     + sh;
      float v3 = s * bf2f(in[base + 57])     + sh;
      float m = fmaxf(fmaxf(v0, v1), fmaxf(v2, v3));
      acc += fmaxf(m, 0.f);
    }
  }
  float val = acc * 0.0625f;
  size_t f = (size_t)b * 1568 + (c * 49 + oy * 7 + ox);
  feat[f]   = val;
  featbf[f] = f2bf(val);
}

// Router: one wave per row. logits = feat @ rw^T + rb, softmax, top-2, renorm.
__global__ __launch_bounds__(256) void router_topk(const float* __restrict__ feat,
                                                   const float* __restrict__ rw,
                                                   const float* __restrict__ rb,
                                                   float* __restrict__ topw,
                                                   int* __restrict__ topi) {
  int gw   = (blockIdx.x * blockDim.x + threadIdx.x) >> 5;  // row b
  int lane = threadIdx.x & 31;
  if (gw >= 512) return;
  const float* f = feat + (size_t)gw * 1568;
  float a0 = 0.f, a1 = 0.f, a2 = 0.f, a3 = 0.f;
  for (int k = lane; k < 1568; k += 32) {
    float fv = f[k];
    a0 += fv * rw[0 * 1568 + k];
    a1 += fv * rw[1 * 1568 + k];
    a2 += fv * rw[2 * 1568 + k];
    a3 += fv * rw[3 * 1568 + k];
  }
  for (int o = 16; o > 0; o >>= 1) {
    a0 += __shfl_xor(a0, o, 32);
    a1 += __shfl_xor(a1, o, 32);
    a2 += __shfl_xor(a2, o, 32);
    a3 += __shfl_xor(a3, o, 32);
  }
  if (lane == 0) {
    float l[4] = {a0 + rb[0], a1 + rb[1], a2 + rb[2], a3 + rb[3]};
    float mx = fmaxf(fmaxf(l[0], l[1]), fmaxf(l[2], l[3]));
    float ex[4], ssum = 0.f;
#pragma unroll
    for (int e = 0; e < 4; ++e) { ex[e] = expf(l[e] - mx); ssum += ex[e]; }
#pragma unroll
    for (int e = 0; e < 4; ++e) ex[e] /= ssum;
    int i0 = 0;
#pragma unroll
    for (int e = 1; e < 4; ++e) if (ex[e] > ex[i0]) i0 = e;
    int i1 = (i0 == 0) ? 1 : 0;
#pragma unroll
    for (int e = 0; e < 4; ++e) if (e != i0 && ex[e] > ex[i1]) i1 = e;
    float tw = ex[i0] + ex[i1];
    topw[2 * gw]     = ex[i0] / tw;
    topw[2 * gw + 1] = ex[i1] / tw;
    topi[2 * gw]     = i0;
    topi[2 * gw + 1] = i1;
  }
}

// ---------------------------------------------------------------------------
// Expert GEMM 1: featbf [512,1568] x w1t[e] [256,1568] -> relu(.+b1) hid bf16
// hid layout [B, E, 256]. grid (32, 4, 4) block 128; wave = one 16x16 tile.
// ---------------------------------------------------------------------------
__global__ __launch_bounds__(128) void expert_gemm1(const bf_t* __restrict__ featbf,
                                                    const bf_t* __restrict__ w1t,
                                                    const float* __restrict__ b1,
                                                    bf_t* __restrict__ hid) {
  int wv = threadIdx.x >> 5, lane = threadIdx.x & 31;
  int g = lane >> 4, hs = lane & 15;
  int mt = blockIdx.x;
  int nt = blockIdx.y * 4 + wv;
  int e  = blockIdx.z;
  const bf_t* A = featbf + (size_t)(mt * 16 + hs) * 1568;
  const bf_t* B = w1t + ((size_t)e * 256 + nt * 16 + hs) * 1568;
  v8f acc = vzero8();
  for (int kk = 0; kk < 1568; kk += 32) {
    __builtin_prefetch(A + kk + 256, 0, 1);   // -> global_prefetch_b8
    __builtin_prefetch(B + kk + 256, 0, 1);
    const unsigned int* pa0 = (const unsigned int*)(A + kk + g * 16);
    const unsigned int* pa1 = (const unsigned int*)(A + kk + g * 16 + 8);
    v8u ap;
    ap[0] = pa0[0]; ap[1] = pa0[1]; ap[2] = pa0[2]; ap[3] = pa0[3];
    ap[4] = pa1[0]; ap[5] = pa1[1]; ap[6] = pa1[2]; ap[7] = pa1[3];
    const unsigned int* pb = (const unsigned int*)(B + kk + g * 16);
    v8u bp;
#pragma unroll
    for (int j = 0; j < 8; ++j) bp[j] = pb[j];
    acc = wmma_bf16(ap, bp, acc);
  }
  int h = nt * 16 + hs;
  float bias = b1[e * 256 + h];
#pragma unroll
  for (int r = 0; r < 8; ++r) {
    int m = mt * 16 + g * 8 + r;
    float v = acc[r] + bias;
    hid[((size_t)m * 4 + e) * 256 + h] = f2bf(v > 0.f ? v : 0.f);
  }
}

// Expert GEMM 2: hid[:,e,:] [512,256] x w2t[e] [128,256] -> all_out f32 + b2
// grid (32, 2, 4) block 128.
__global__ __launch_bounds__(128) void expert_gemm2(const bf_t* __restrict__ hid,
                                                    const bf_t* __restrict__ w2t,
                                                    const float* __restrict__ b2,
                                                    float* __restrict__ allo) {
  int wv = threadIdx.x >> 5, lane = threadIdx.x & 31;
  int g = lane >> 4, hs = lane & 15;
  int mt = blockIdx.x;
  int nt = blockIdx.y * 4 + wv;
  int e  = blockIdx.z;
  const bf_t* A = hid + ((size_t)(mt * 16 + hs) * 4 + e) * 256;
  const bf_t* B = w2t + ((size_t)e * 128 + nt * 16 + hs) * 256;
  v8f acc = vzero8();
#pragma unroll
  for (int kk = 0; kk < 256; kk += 32) {
    const unsigned int* pa0 = (const unsigned int*)(A + kk + g * 16);
    const unsigned int* pa1 = (const unsigned int*)(A + kk + g * 16 + 8);
    v8u ap;
    ap[0] = pa0[0]; ap[1] = pa0[1]; ap[2] = pa0[2]; ap[3] = pa0[3];
    ap[4] = pa1[0]; ap[5] = pa1[1]; ap[6] = pa1[2]; ap[7] = pa1[3];
    const unsigned int* pb = (const unsigned int*)(B + kk + g * 16);
    v8u bp;
#pragma unroll
    for (int j = 0; j < 8; ++j) bp[j] = pb[j];
    acc = wmma_bf16(ap, bp, acc);
  }
  int d = nt * 16 + hs;
  float bias = b2[e * 128 + d];
#pragma unroll
  for (int r = 0; r < 8; ++r) {
    int m = mt * 16 + g * 8 + r;
    allo[((size_t)m * 4 + e) * 128 + d] = acc[r] + bias;
  }
}

// Top-2 weighted sum + L2 normalize. One block per row, 128 threads (= D).
__global__ __launch_bounds__(128) void finalize_emb(const float* __restrict__ allo,
                                                    const float* __restrict__ topw,
                                                    const int* __restrict__ topi,
                                                    float* __restrict__ out) {
  int b = blockIdx.x, d = threadIdx.x;
  int i0 = topi[2 * b], i1 = topi[2 * b + 1];
  float w0 = topw[2 * b], w1 = topw[2 * b + 1];
  float v = w0 * allo[((size_t)b * 4 + i0) * 128 + d] +
            w1 * allo[((size_t)b * 4 + i1) * 128 + d];
  __shared__ float red[128];
  red[d] = v * v;
  __syncthreads();
  for (int s = 64; s > 0; s >>= 1) {
    if (d < s) red[d] += red[d + s];
    __syncthreads();
  }
  float norm = sqrtf(red[0]);
  out[(size_t)b * 128 + d] = v / fmaxf(norm, 1e-12f);
}

// ===========================================================================
extern "C" void kernel_launch(void* const* d_in, const int* in_sizes, int n_in,
                              void* d_out, int out_size, void* d_ws, size_t ws_size,
                              hipStream_t stream) {
  (void)in_sizes; (void)n_in; (void)out_size; (void)ws_size;
  const float* x        = (const float*)d_in[0];
  const float* conv1_w  = (const float*)d_in[1];
  const float* conv1_b  = (const float*)d_in[2];
  const float* bn1_g    = (const float*)d_in[3];
  const float* bn1_b    = (const float*)d_in[4];
  const float* conv2_w  = (const float*)d_in[5];
  const float* conv2_b  = (const float*)d_in[6];
  const float* bn2_g    = (const float*)d_in[7];
  const float* bn2_b    = (const float*)d_in[8];
  const float* router_w = (const float*)d_in[9];
  const float* router_b = (const float*)d_in[10];
  const float* w1       = (const float*)d_in[11];
  const float* b1       = (const float*)d_in[12];
  const float* w2       = (const float*)d_in[13];
  const float* b2       = (const float*)d_in[14];
  float* out = (float*)d_out;

  char* base = (char*)d_ws;
  size_t off = 0;
  auto alloc = [&](size_t bytes) -> void* {
    void* p = base + off;
    off += (bytes + 255) & ~(size_t)255;
    return p;
  };
  bf_t*  c1out  = (bf_t*)alloc((size_t)512 * 16 * 12544 * 2);  // conv1 out (+bias)
  bf_t*  h1     = (bf_t*)alloc((size_t)512 * 16 * 3136 * 2);   // pooled stage1
  bf_t*  c2out  = (bf_t*)alloc((size_t)512 * 32 * 3136 * 2);   // conv2 out (+bias)
  float* feat   = (float*)alloc((size_t)512 * 1568 * 4);
  bf_t*  featbf = (bf_t*)alloc((size_t)512 * 1568 * 2);
  bf_t*  w1t    = (bf_t*)alloc((size_t)4 * 256 * 1568 * 2);
  bf_t*  w2t    = (bf_t*)alloc((size_t)4 * 128 * 256 * 2);
  bf_t*  wc1t   = (bf_t*)alloc((size_t)16 * 96 * 2);
  bf_t*  wc2t   = (bf_t*)alloc((size_t)32 * 416 * 2);
  float* p1s    = (float*)alloc((size_t)16 * 128 * 4);
  float* p1q    = (float*)alloc((size_t)16 * 128 * 4);
  float* p2s    = (float*)alloc((size_t)32 * 128 * 4);
  float* p2q    = (float*)alloc((size_t)32 * 128 * 4);
  float* bnp1   = (float*)alloc(32 * 4);
  float* bnp2   = (float*)alloc(64 * 4);
  float* topw   = (float*)alloc(512 * 2 * 4);
  int*   topi   = (int*)alloc(512 * 2 * 4);
  bf_t*  hid    = (bf_t*)alloc((size_t)512 * 4 * 256 * 2);
  float* allo   = (float*)alloc((size_t)512 * 4 * 128 * 4);

  // weight prep
  pack_pad_bf16<<<(16 * 96 + 255) / 256, 256, 0, stream>>>(conv1_w, wc1t, 16, 75, 96);
  pack_pad_bf16<<<(32 * 416 + 255) / 256, 256, 0, stream>>>(conv2_w, wc2t, 32, 400, 416);
  transpose_to_bf16<<<(4 * 1568 * 256 + 255) / 256, 256, 0, stream>>>(w1, w1t, 4, 1568, 256);
  transpose_to_bf16<<<(4 * 256 * 128 + 255) / 256, 256, 0, stream>>>(w2, w2t, 4, 256, 128);

  // stage 1: conv1 (WMMA) -> BN stats -> BN+ReLU+maxpool
  conv5x5_wmma<3, 3, 112, 1, true><<<dim3(196, 512), 128, 0, stream>>>(x, wc1t, conv1_b, c1out);
  chan_stats<<<dim3(16, 128), 256, 0, stream>>>(c1out, p1s, p1q, 16, 12544, 512 * 12544);
  bn_finalize<<<1, 32, 0, stream>>>(p1s, p1q, 128, bn1_g, bn1_b, bnp1, 16,
                                    1.f / (512.f * 12544.f));
  bnpool<<<(512 * 16 * 3136) / 256, 256, 0, stream>>>(c1out, bnp1, h1, 16, 112,
                                                      512 * 16 * 3136);

  // stage 2: conv2 (WMMA) -> BN stats -> BN+ReLU+maxpool+avgpool -> feat
  conv5x5_wmma<16, 13, 56, 2, false><<<dim3(49, 512), 128, 0, stream>>>(h1, wc2t, conv2_b, c2out);
  chan_stats<<<dim3(32, 128), 256, 0, stream>>>(c2out, p2s, p2q, 32, 3136, 512 * 3136);
  bn_finalize<<<1, 32, 0, stream>>>(p2s, p2q, 128, bn2_g, bn2_b, bnp2, 32,
                                    1.f / (512.f * 3136.f));
  bnpoolavg<<<(512 * 32 * 49) / 256, 256, 0, stream>>>(c2out, bnp2, feat, featbf);

  // router + experts (WMMA) + finalize
  router_topk<<<64, 256, 0, stream>>>(feat, router_w, router_b, topw, topi);
  expert_gemm1<<<dim3(32, 4, 4), 128, 0, stream>>>(featbf, w1t, b1, hid);
  expert_gemm2<<<dim3(32, 2, 4), 128, 0, stream>>>(hid, w2t, b2, allo);
  finalize_emb<<<512, 128, 0, stream>>>(allo, topw, topi, out);
}
